// M3oE_4844723110492
// MI455X (gfx1250) — matrix operand order repaired and checked by reading, not compiled, not run on hardware
//
#include <hip/hip_runtime.h>
#include <hip/hip_bf16.h>
#include <stdint.h>

#define F_  26
#define V_  100000
#define DK_ 16
#define D_  4
#define E_  8
#define H1_ 128
#define H2_ 64
#define B_  16384
#define IN_ 416   // F_*DK_
#define MT_ 32    // rows per block

typedef __attribute__((ext_vector_type(16))) __bf16 v16bf;
typedef __attribute__((ext_vector_type(8)))  float  v8f;
typedef __attribute__((ext_vector_type(4)))  uint32_t u32x4;
typedef __attribute__((ext_vector_type(8)))  int      i32x8;
typedef __attribute__((ext_vector_type(4)))  int      i32x4;

union Frag {
    v16bf v;
    uint4 u[2];
};

__device__ __forceinline__ uint16_t f2bf(float f) {
    uint32_t u = __builtin_bit_cast(uint32_t, f);
    u += 0x7FFFu + ((u >> 16) & 1u);   // round-to-nearest-even
    return (uint16_t)(u >> 16);
}

// ---------------- prep kernels ----------------

__global__ void gather_x_kernel(const int* __restrict__ feats,
                                const float* __restrict__ emb,
                                uint16_t* __restrict__ xb) {
    int i = blockIdx.x * blockDim.x + threadIdx.x;
    if (i >= B_ * F_) return;
    int b = i / F_, f = i % F_;
    int idx = feats[i];
    const float4* s = (const float4*)(emb + ((size_t)f * V_ + (size_t)idx) * DK_);
    uint32_t o[8];
#pragma unroll
    for (int j = 0; j < 4; ++j) {
        float4 t = s[j];
        o[2 * j]     = (uint32_t)f2bf(t.x) | ((uint32_t)f2bf(t.y) << 16);
        o[2 * j + 1] = (uint32_t)f2bf(t.z) | ((uint32_t)f2bf(t.w) << 16);
    }
    uint4* d = (uint4*)(xb + (size_t)b * IN_ + f * DK_);
    d[0] = make_uint4(o[0], o[1], o[2], o[3]);
    d[1] = make_uint4(o[4], o[5], o[6], o[7]);
}

__global__ void conv_w1_kernel(const float* __restrict__ W1, uint16_t* __restrict__ w1t) {
    int i = blockIdx.x * blockDim.x + threadIdx.x;
    if (i >= E_ * IN_ * H1_) return;
    int e = i / (IN_ * H1_), r = i % (IN_ * H1_);
    int k = r / H1_, n = r % H1_;
    w1t[((size_t)e * H1_ + n) * IN_ + k] = f2bf(W1[i]);
}

__global__ void conv_w2_kernel(const float* __restrict__ W2, uint16_t* __restrict__ w2t) {
    int i = blockIdx.x * blockDim.x + threadIdx.x;
    if (i >= E_ * H1_ * H2_) return;
    int e = i / (H1_ * H2_), r = i % (H1_ * H2_);
    int k = r / H2_, n = r % H2_;
    w2t[((size_t)e * H2_ + n) * H1_ + k] = f2bf(W2[i]);
}

__global__ void conv_wg_kernel(const float* __restrict__ Wg, uint16_t* __restrict__ wgt) {
    int i = blockIdx.x * blockDim.x + threadIdx.x;
    if (i >= D_ * E_ * IN_) return;
    int n = i / IN_, k = i % IN_;
    int d = n / E_, e = n % E_;
    wgt[i] = f2bf(Wg[((size_t)d * IN_ + k) * E_ + e]);
}

// ---------------- fused main kernel ----------------
__global__ __launch_bounds__(256, 1) void m3oe_main_kernel(
    const uint16_t* __restrict__ xb,  const uint16_t* __restrict__ w1t,
    const uint16_t* __restrict__ w2t, const uint16_t* __restrict__ wgt,
    const float* __restrict__ b1, const float* __restrict__ b2,
    const float* __restrict__ bg,
    const float* __restrict__ Wo, const float* __restrict__ bo,
    const int* __restrict__ dom, float* __restrict__ out)
{
    __shared__ __align__(16) uint16_t xs[MT_][IN_];         // TDM-staged x tile
    __shared__ __align__(16) uint16_t h1s[E_][16][H1_ + 8]; // per-wave h1 staging
    __shared__ float gall[MT_][32];                         // biased gate logits
    __shared__ float fin[MT_][H2_];                         // mixed expert outputs

    const int wave  = threadIdx.x >> 5;
    const int lane  = threadIdx.x & 31;
    const int e     = wave;
    const int b0    = blockIdx.x * MT_;
    const int ncol  = lane & 15;
    const int hi    = lane >> 4;
    const int koffA = hi * 8;
    const int koffB = hi * 8;

    for (int i = threadIdx.x; i < MT_ * H2_; i += 256)
        (&fin[0][0])[i] = 0.0f;

    // ---- TDM: DMA the 32x416 bf16 x tile into LDS ----
    if (wave == 0) {
        const uint64_t ga  = (uint64_t)(uintptr_t)(xb + (size_t)b0 * IN_);
        const uint32_t lds = (uint32_t)(uintptr_t)&xs[0][0];
        u32x4 g0;
        g0.x = 1u;                                             // count=1
        g0.y = lds;                                            // lds_addr
        g0.z = (uint32_t)ga;                                   // global_addr[31:0]
        g0.w = ((uint32_t)(ga >> 32) & 0x1FFFFFFu) | (2u << 30); // addr[56:32]|type=2
        i32x8 g1;
        g1[0] = (int)(1u << 16);                               // data_size=2B, mask=0
        g1[1] = (int)((IN_ & 0xFFFF) << 16);                   // tensor_dim0[15:0]
        g1[2] = (int)(((IN_ >> 16) & 0xFFFF) | ((MT_ & 0xFFFF) << 16)); // td0 hi|td1 lo
        g1[3] = (int)(((MT_ >> 16) & 0xFFFF) | ((IN_ & 0xFFFF) << 16)); // td1 hi|tile_dim0
        g1[4] = MT_;                                           // tile_dim1=32, tile_dim2=0
        g1[5] = IN_;                                           // dim0_stride[31:0]
        g1[6] = (int)((IN_ & 0xFFFF) << 16);                   // dim0_str hi|dim1_str lo
        g1[7] = 0;                                             // dim1_stride hi
        const i32x4 gz4 = {0, 0, 0, 0};
        const i32x8 gz8 = {0, 0, 0, 0, 0, 0, 0, 0};
        __builtin_amdgcn_tensor_load_to_lds(g0, g1, gz4, gz4, gz8, 0);
    }
    __builtin_amdgcn_s_wait_tensorcnt(0);
    __syncthreads();

    const v8f vzero = {0.f, 0.f, 0.f, 0.f, 0.f, 0.f, 0.f, 0.f};

    const uint16_t* xr0 = &xs[ncol][0];       // sub-tile 0 A rows (LDS)
    const uint16_t* xr1 = &xs[16 + ncol][0];  // sub-tile 1 A rows (LDS)

    // ---- gate logits via WMMA: waves 0..3 compute one 16x16 tile each ----
    if (wave < 4) {
        const int s  = wave >> 1;
        const int nt = wave & 1;
        const uint16_t* xr = (s == 0) ? xr0 : xr1;
        v8f gacc = vzero;
        for (int ks = 0; ks < 13; ++ks) {
            const int k0 = ks * 32;
            Frag a, bm;
            a.u[0] = *(const uint4*)(xr + k0 + koffA);
            a.u[1] = *(const uint4*)(xr + k0 + koffA + 16);
            const uint16_t* bp = wgt + (size_t)(nt * 16 + ncol) * IN_ + k0 + koffB;
            bm.u[0] = *(const uint4*)bp;
            bm.u[1] = *(const uint4*)(bp + 16);
            gacc = __builtin_amdgcn_wmma_f32_16x16x32_bf16(
                false, a.v, false, bm.v, (short)0, gacc, false, false);
        }
        const float bgv = bg[nt * 16 + ncol];
#pragma unroll
        for (int v = 0; v < 8; ++v)
            gall[s * 16 + v + 8 * hi][nt * 16 + ncol] = gacc[v] + bgv;
    }

    // ---- GEMM1: x(32x416) * W1[e](416x128), software-pipelined ----
    v8f acc[2][8];
#pragma unroll
    for (int s = 0; s < 2; ++s)
#pragma unroll
        for (int t = 0; t < 8; ++t) acc[s][t] = vzero;

    Frag a0, a1;
    a0.u[0] = *(const uint4*)(xr0 + koffA);
    a0.u[1] = *(const uint4*)(xr0 + koffA + 16);
    a1.u[0] = *(const uint4*)(xr1 + koffA);
    a1.u[1] = *(const uint4*)(xr1 + koffA + 16);

    const uint16_t* wbase = w1t + (size_t)(e * H1_ + ncol) * IN_ + koffB;
    for (int ks = 0; ks < 13; ++ks) {
        const int k0 = ks * 32;
        Frag na0, na1;
        if (ks < 12) {
            na0.u[0] = *(const uint4*)(xr0 + k0 + 32 + koffA);
            na0.u[1] = *(const uint4*)(xr0 + k0 + 32 + koffA + 16);
            na1.u[0] = *(const uint4*)(xr1 + k0 + 32 + koffA);
            na1.u[1] = *(const uint4*)(xr1 + k0 + 32 + koffA + 16);
            __builtin_prefetch((const void*)(wbase + k0 + 32), 0, 3);
        }
        Frag bcur;
        bcur.u[0] = *(const uint4*)(wbase + k0);
        bcur.u[1] = *(const uint4*)(wbase + k0 + 16);
#pragma unroll
        for (int t = 0; t < 8; ++t) {
            Frag bnxt;
            if (t < 7) {
                const uint16_t* wn = wbase + (size_t)(t + 1) * 16 * IN_ + k0;
                bnxt.u[0] = *(const uint4*)wn;
                bnxt.u[1] = *(const uint4*)(wn + 16);
            }
            acc[0][t] = __builtin_amdgcn_wmma_f32_16x16x32_bf16(
                false, a0.v, false, bcur.v, (short)0, acc[0][t], false, false);
            acc[1][t] = __builtin_amdgcn_wmma_f32_16x16x32_bf16(
                false, a1.v, false, bcur.v, (short)0, acc[1][t], false, false);
            if (t < 7) bcur = bnxt;
        }
        if (ks < 12) { a0 = na0; a1 = na1; }
    }

    // ---- per sub-tile: bias+relu -> LDS bf16 -> GEMM2 ----
    v8f acc2[2][4];
#pragma unroll
    for (int s = 0; s < 2; ++s) {
#pragma unroll
        for (int t = 0; t < 8; ++t) {
            const float bias = b1[e * H1_ + t * 16 + ncol];
#pragma unroll
            for (int v = 0; v < 8; ++v) {
                float h = acc[s][t][v] + bias;
                h = h > 0.0f ? h : 0.0f;
                h1s[e][v + 8 * hi][t * 16 + ncol] = f2bf(h);
            }
        }
#pragma unroll
        for (int t = 0; t < 4; ++t) acc2[s][t] = vzero;
        for (int ks = 0; ks < 4; ++ks) {
            const int k0 = ks * 32;
            Frag a;
            const uint16_t* ap = &h1s[e][ncol][k0 + koffA];
            a.u[0] = *(const uint4*)ap;
            a.u[1] = *(const uint4*)(ap + 16);
#pragma unroll
            for (int t = 0; t < 4; ++t) {
                const uint16_t* bp =
                    w2t + (size_t)(e * H2_ + t * 16 + ncol) * H1_ + k0 + koffB;
                Frag bm;
                bm.u[0] = *(const uint4*)bp;
                bm.u[1] = *(const uint4*)(bp + 16);
                acc2[s][t] = __builtin_amdgcn_wmma_f32_16x16x32_bf16(
                    false, a.v, false, bm.v, (short)0, acc2[s][t], false, false);
            }
        }
#pragma unroll
        for (int t = 0; t < 4; ++t) {
            const float bias = b2[e * H2_ + t * 16 + ncol];
#pragma unroll
            for (int v = 0; v < 8; ++v) {
                float h = acc2[s][t][v] + bias;
                acc2[s][t][v] = h > 0.0f ? h : 0.0f;
            }
        }
    }

    __syncthreads();   // gall ready

    // ---- softmax over experts (per row), pick my expert's gate ----
    float gv[2][8];
#pragma unroll
    for (int s = 0; s < 2; ++s) {
#pragma unroll
        for (int v = 0; v < 8; ++v) {
            const int m = s * 16 + 8 * hi + v;
            const int base = dom[b0 + m] * E_;
            float mx = gall[m][base];
#pragma unroll
            for (int e2 = 1; e2 < E_; ++e2) mx = fmaxf(mx, gall[m][base + e2]);
            float den = 0.0f, mine = 0.0f;
#pragma unroll
            for (int e2 = 0; e2 < E_; ++e2) {
                float t = __expf(gall[m][base + e2] - mx);
                den += t;
                if (e2 == e) mine = t;
            }
            gv[s][v] = mine / den;
        }
    }

    // ---- gate-weighted mixture into shared accumulator ----
#pragma unroll
    for (int s = 0; s < 2; ++s)
#pragma unroll
        for (int t = 0; t < 4; ++t)
#pragma unroll
            for (int v = 0; v < 8; ++v)
                atomicAdd(&fin[s * 16 + 8 * hi + v][t * 16 + ncol],
                          gv[s][v] * acc2[s][t][v]);
    __syncthreads();

    // ---- final 64->1 projection ----
    if (threadIdx.x < MT_) {
        float s = bo[0];
        for (int n = 0; n < H2_; ++n) s += fin[threadIdx.x][n] * Wo[n];
        out[b0 + threadIdx.x] = s;
    }
}

// ---------------- launcher ----------------
extern "C" void kernel_launch(void* const* d_in, const int* in_sizes, int n_in,
                              void* d_out, int out_size, void* d_ws, size_t ws_size,
                              hipStream_t stream) {
    const int*   feats = (const int*)d_in[0];
    const int*   dom   = (const int*)d_in[1];
    const float* emb   = (const float*)d_in[2];
    const float* W1    = (const float*)d_in[3];
    const float* b1    = (const float*)d_in[4];
    const float* W2    = (const float*)d_in[5];
    const float* b2    = (const float*)d_in[6];
    const float* Wg    = (const float*)d_in[7];
    const float* bg    = (const float*)d_in[8];
    const float* Wo    = (const float*)d_in[9];
    const float* bo    = (const float*)d_in[10];
    float* out = (float*)d_out;

    uint16_t* xb  = (uint16_t*)d_ws;                    // B*IN bf16
    uint16_t* w1t = xb  + (size_t)B_ * IN_;             // E*H1*IN bf16
    uint16_t* w2t = w1t + (size_t)E_ * H1_ * IN_;       // E*H2*H1 bf16
    uint16_t* wgt = w2t + (size_t)E_ * H2_ * H1_;       // D*E*IN bf16

    gather_x_kernel<<<(B_ * F_ + 255) / 256, 256, 0, stream>>>(feats, emb, xb);
    conv_w1_kernel<<<(E_ * IN_ * H1_ + 255) / 256, 256, 0, stream>>>(W1, w1t);
    conv_w2_kernel<<<(E_ * H1_ * H2_ + 255) / 256, 256, 0, stream>>>(W2, w2t);
    conv_wg_kernel<<<(D_ * E_ * IN_ + 255) / 256, 256, 0, stream>>>(Wg, wgt);

    m3oe_main_kernel<<<B_ / MT_, 256, 0, stream>>>(xb, w1t, w2t, wgt, b1, b2,
                                                   bg, Wo, bo, dom, out);
}
